// RPN_44263932952808
// MI455X (gfx1250) — matrix operand order repaired and checked by reading, not compiled, not run on hardware
//
#include <hip/hip_runtime.h>

typedef _Float16 half_t;
typedef __attribute__((ext_vector_type(16))) _Float16 v16h;
typedef __attribute__((ext_vector_type(8)))  _Float16 v8h;
typedef __attribute__((ext_vector_type(8)))  float    v8f;

#define CIN      256
#define CMID     512
#define KTOT     2304      // 256 * 9
#define KBLKS    72        // KTOT / 32
#define PRE_NMS  6000
#define POST_NMS 1000
#define NPAD     65536
#define FPAD     8192
#define PTOT     21824     // sum of H*W over levels
#define ANCHB    65472     // PTOT * 3
#define NSEG     10        // B * levels
#define NMS_THR  0.7f

__constant__ int c_ktab[5]     = {6000, 6000, 3072, 768, 192};
__constant__ int c_baseanch[5] = {0, 49152, 61440, 64512, 65280};

#define WMMA_F16(A, B, C) __builtin_amdgcn_wmma_f32_16x16x32_f16( \
        false, (A), false, (B), (short)0, (C), false, false)

// ---------------------------------------------------------------------------
// 0) NCHW f32 -> NHWC f16 with 1-pixel zero halo:
//    featc[b][row=(h+1)][col=(w+1)][c], rows = (H+2)*(W+2)
// ---------------------------------------------------------------------------
__global__ void k_nhwc(const float* __restrict__ feat, half_t* __restrict__ featc,
                       int H, int W) {
    int c   = threadIdx.x;                 // 256 threads = channels
    int row = blockIdx.x;                  // (H+2)*(W+2)
    int b   = blockIdx.y;
    int W2  = W + 2;
    int nrows = (H + 2) * W2;
    int h = row / W2 - 1;                  // uniform per block -> scalar div
    int w = row % W2 - 1;
    float v = 0.0f;
    if (h >= 0 && h < H && w >= 0 && w < W)
        v = feat[(((size_t)b * CIN + c) * H + h) * W + w];
    featc[(((size_t)b * nrows + row) << 8) + c] = (half_t)v;
}

// ---------------------------------------------------------------------------
// 1) Repack conv weights (512,256,3,3) f32 -> WMMA B-fragment-ordered f16.
//    GEMM K-order is tap-major: K = q*256 + c  (q = kh*3+kw).
//    wpack[kblk][nblk16][lane][e], lane = r*16+n, element e -> K = kblk*32+8r+(e&7)+16*(e>>3)
// ---------------------------------------------------------------------------
__global__ void k_pack_w(const float* __restrict__ w_conv, half_t* __restrict__ wpack) {
    int tid = blockIdx.x * blockDim.x + threadIdx.x;   // KBLKS*32*32*16 = 1179648
    int e    = tid & 15;
    int lane = (tid >> 4) & 31;
    int nblk = (tid >> 9) & 31;
    int kblk = tid >> 14;
    if (kblk >= KBLKS) return;
    int o = nblk * 16 + (lane & 15);
    int K = kblk * 32 + (lane >> 4) * 8 + (e & 7) + (e >> 3) * 16;
    int q = K >> 8;          // tap index kh*3+kw
    int c = K & 255;         // channel
    wpack[tid] = (half_t)w_conv[(size_t)o * KTOT + c * 9 + q];
}

// ---------------------------------------------------------------------------
// 2) 3x3 conv 256->512 + bias + ReLU via implicit-GEMM WMMA f32_16x16x32_f16.
//    Block = 256 threads (8 waves); tile = 16 pixels (M) x 512 outs (N).
//    Each wave owns 64 N (4 accumulators). No LDS, no barriers: A fragments
//    are contiguous 16B loads from the haloed NHWC f16 feature buffer.
// ---------------------------------------------------------------------------
__global__ __launch_bounds__(256)
void k_conv(const half_t* __restrict__ featc, const half_t* __restrict__ wpack,
            const float* __restrict__ bconv, half_t* __restrict__ hidden,
            int logW, int basepix) {
    const int W    = 1 << logW;
    const int W2   = W + 2;
    const int nrows = (W + 2) * (W + 2);   // H == W on every level
    const int b    = blockIdx.y;
    const int p0   = blockIdx.x * 16;
    const int lane = threadIdx.x & 31;
    const int wv   = threadIdx.x >> 5;
    const int r    = lane >> 4;
    const int mrow = lane & 15;
    const int nb0  = wv * 4;               // first of 4 owned 16-wide N tiles

    // pixel this lane's A-row corresponds to
    const int p  = p0 + mrow;
    const int ph = p >> logW;
    const int pw = p & (W - 1);

    v8f acc0 = {}, acc1 = {}, acc2 = {}, acc3 = {};

    for (int q = 0; q < 9; ++q) {
        const int kh = q / 3, kw = q % 3;  // tiny scalar math, 9 iters
        const half_t* arow = featc +
            ((((size_t)b * nrows) + (size_t)(ph + kh) * W2 + (pw + kw)) << 8);
        if (q < 8) {
            const int qn = q + 1;
            const half_t* arown = featc +
                ((((size_t)b * nrows) + (size_t)(ph + qn / 3) * W2 + (pw + qn % 3)) << 8);
            __builtin_prefetch(arown, 0, 0);   // global_prefetch_b8
        }
#pragma unroll
        for (int cb = 0; cb < 8; ++cb) {
            const int kblk = q * 8 + cb;
            // A fragment: two contiguous 16B chunks of consecutive channels
            v8h a0 = *(const v8h*)(arow + cb * 32 + 8 * r);        // K = 8r..8r+7
            v8h a1 = *(const v8h*)(arow + cb * 32 + 16 + 8 * r);   // K = 16+8r..
            v16h a;
#pragma unroll
            for (int e = 0; e < 8; ++e) { a[e] = a0[e]; a[8 + e] = a1[e]; }
            const half_t* wp = wpack + ((((size_t)kblk * 32 + nb0) * 32 + lane) << 4);
            v16h b0 = *(const v16h*)(wp);
            v16h b1 = *(const v16h*)(wp + 512);
            v16h b2 = *(const v16h*)(wp + 1024);
            v16h b3 = *(const v16h*)(wp + 1536);
            acc0 = WMMA_F16(a, b0, acc0);
            acc1 = WMMA_F16(a, b1, acc1);
            acc2 = WMMA_F16(a, b2, acc2);
            acc3 = WMMA_F16(a, b3, acc3);
        }
    }
    // epilogue: C layout lane=(r,n), VGPR v -> M = v + 8r
    const int n = lane & 15;
#pragma unroll
    for (int j = 0; j < 4; ++j) {
        v8f acc = (j == 0) ? acc0 : (j == 1) ? acc1 : (j == 2) ? acc2 : acc3;
        int o = (nb0 + j) * 16 + n;
        float bias = bconv[o];
#pragma unroll
        for (int v = 0; v < 8; ++v) {
            int M = v + 8 * r;
            float y = acc[v] + bias;
            y = y > 0.0f ? y : 0.0f;
            hidden[((size_t)(b * PTOT + basepix + p0 + M)) * CMID + o] = (half_t)y;
        }
    }
}

// ---------------------------------------------------------------------------
// 3) 1x1 heads + sigmoid + anchor decode + clip; fill padded sort segments
// ---------------------------------------------------------------------------
__global__ void k_head(const half_t* __restrict__ hidden,
                       const float* __restrict__ w_cls, const float* __restrict__ b_cls,
                       const float* __restrict__ w_reg, const float* __restrict__ b_reg,
                       const float* __restrict__ img_info,
                       float* __restrict__ skey, int* __restrict__ sidx,
                       float* __restrict__ boxes,
                       int level, int H, int W, int basepix, int baseanch,
                       int strideL, float sizeL) {
    int tid = blockIdx.x * blockDim.x + threadIdx.x;   // 2 * 65536
    int b = tid >> 16;
    int i = tid & (NPAD - 1);
    if (b >= 2) return;
    int seg = b * 5 + level;
    int nanch = H * W * 3;
    if (i >= nanch) {
        skey[(size_t)seg * NPAD + i] = -__builtin_inff();
        sidx[(size_t)seg * NPAD + i] = i;
        return;
    }
    int p = i / 3, a = i % 3;
    const half_t* hrow = hidden + ((size_t)(b * PTOT + basepix + p)) * CMID;
    float dc  = b_cls[a];
    float dr0 = b_reg[a * 4 + 0], dr1 = b_reg[a * 4 + 1];
    float dr2 = b_reg[a * 4 + 2], dr3 = b_reg[a * 4 + 3];
    const float* wc = w_cls + (size_t)a * CMID;
    const float* wr = w_reg + (size_t)a * 4 * CMID;
    for (int o = 0; o < CMID; o += 8) {
        v8h hv8 = *(const v8h*)(hrow + o);
#pragma unroll
        for (int e = 0; e < 8; ++e) {
            float hv = (float)hv8[e];
            dc  += hv * wc[o + e];
            dr0 += hv * wr[o + e];
            dr1 += hv * wr[CMID + o + e];
            dr2 += hv * wr[2 * CMID + o + e];
            dr3 += hv * wr[3 * CMID + o + e];
        }
    }
    float prob = 1.0f / (1.0f + expf(-dc));
    int ph = p / W, pw = p % W;
    float cx = (pw + 0.5f) * (float)strideL;
    float cy = (ph + 0.5f) * (float)strideL;
    const float ratios[3] = {0.5f, 1.0f, 2.0f};
    float sr = sqrtf(ratios[a]);
    float ws = sizeL / sr, hs = sizeL * sr;
    float ncx = cx + dr0 * ws;
    float ncy = cy + dr1 * hs;
    float nw  = ws * expf(dr2);
    float nh  = hs * expf(dr3);
    float imH = img_info[b * 2 + 0], imW = img_info[b * 2 + 1];
    float x1 = fminf(fmaxf(ncx - 0.5f * nw, 0.0f), imW);
    float y1 = fminf(fmaxf(ncy - 0.5f * nh, 0.0f), imH);
    float x2 = fminf(fmaxf(ncx + 0.5f * nw, 0.0f), imW);
    float y2 = fminf(fmaxf(ncy + 0.5f * nh, 0.0f), imH);
    float* bp = boxes + ((size_t)b * ANCHB + baseanch + i) * 4;
    bp[0] = x1; bp[1] = y1; bp[2] = x2; bp[3] = y2;
    skey[(size_t)seg * NPAD + i] = prob;
    sidx[(size_t)seg * NPAD + i] = i;
}

// ---------------------------------------------------------------------------
// 4) generic bitonic compare-swap pass, descending, segmented
// ---------------------------------------------------------------------------
__global__ void k_bitonic(float* __restrict__ key, int* __restrict__ idx,
                          int len, int kk, int j, int nseg) {
    int gid = blockIdx.x * blockDim.x + threadIdx.x;
    int seg = gid / len;
    int i   = gid % len;
    if (seg >= nseg) return;
    int ixj = i ^ j;
    if (ixj <= i) return;
    float* K = key + (size_t)seg * len;
    int*   I = idx + (size_t)seg * len;
    float a = K[i], b = K[ixj];
    bool desc = ((i & kk) == 0);
    bool sw = desc ? (a < b) : (a > b);
    if (sw) {
        K[i] = b; K[ixj] = a;
        int t = I[i]; I[i] = I[ixj]; I[ixj] = t;
    }
}

// ---------------------------------------------------------------------------
// 5) gather top-k boxes/scores per segment
// ---------------------------------------------------------------------------
__global__ void k_gather(const float* __restrict__ skey, const int* __restrict__ sidx,
                         const float* __restrict__ boxes,
                         float* __restrict__ sctop, float4* __restrict__ bxtop) {
    int tid = blockIdx.x * blockDim.x + threadIdx.x;   // NSEG * PRE_NMS
    int seg = tid / PRE_NMS, i = tid % PRE_NMS;
    if (seg >= NSEG) return;
    int l = seg % 5, b = seg / 5;
    int k = c_ktab[l];
    if (i >= k) {
        sctop[seg * PRE_NMS + i] = -__builtin_inff();
        bxtop[seg * PRE_NMS + i] = make_float4(0.f, 0.f, 0.f, 0.f);
        return;
    }
    int si = sidx[(size_t)seg * NPAD + i];
    const float* bp = boxes + ((size_t)b * ANCHB + c_baseanch[l] + si) * 4;
    sctop[seg * PRE_NMS + i] = skey[(size_t)seg * NPAD + i];
    bxtop[seg * PRE_NMS + i] = make_float4(bp[0], bp[1], bp[2], bp[3]);
}

// ---------------------------------------------------------------------------
// 6) greedy sequential NMS, one block per segment
// ---------------------------------------------------------------------------
__global__ void k_nms(const float4* __restrict__ bxtop, int* __restrict__ keepout) {
    __shared__ unsigned char keep[PRE_NMS];
    __shared__ int sup;
    int seg = blockIdx.x;
    int k = c_ktab[seg % 5];
    const float4* B = bxtop + (size_t)seg * PRE_NMS;
    for (int i = threadIdx.x; i < PRE_NMS; i += blockDim.x) keep[i] = 1;
    __syncthreads();
    for (int i = 1; i < k; ++i) {
        if (threadIdx.x == 0) sup = 0;
        __syncthreads();
        float4 bi = B[i];
        float ai = (bi.z - bi.x) * (bi.w - bi.y);
        for (int j = threadIdx.x; j < i; j += blockDim.x) {
            if (keep[j]) {
                float4 bj = B[j];
                float xx1 = fmaxf(bi.x, bj.x), yy1 = fmaxf(bi.y, bj.y);
                float xx2 = fminf(bi.z, bj.z), yy2 = fminf(bi.w, bj.w);
                float w = fmaxf(xx2 - xx1, 0.0f), h = fmaxf(yy2 - yy1, 0.0f);
                float inter = w * h;
                float aj = (bj.z - bj.x) * (bj.w - bj.y);
                float iou = inter / (ai + aj - inter + 1e-8f);
                if (iou > NMS_THR) atomicOr(&sup, 1);
            }
        }
        __syncthreads();
        if (threadIdx.x == 0 && sup) keep[i] = 0;
        __syncthreads();
    }
    for (int i = threadIdx.x; i < PRE_NMS; i += blockDim.x)
        keepout[(size_t)seg * PRE_NMS + i] = (i < k) ? keep[i] : 0;
}

// ---------------------------------------------------------------------------
// 7) compact kept entries (already sorted desc) -> 1000/segment, -inf pads
// ---------------------------------------------------------------------------
__global__ void k_compact(const float* __restrict__ sctop, const float4* __restrict__ bxtop,
                          const int* __restrict__ keep,
                          float* __restrict__ s2, float4* __restrict__ b2) {
    int seg = blockIdx.x;
    if (threadIdx.x != 0) return;
    int k = c_ktab[seg % 5];
    int cnt = 0;
    for (int i = 0; i < k && cnt < POST_NMS; ++i) {
        if (keep[(size_t)seg * PRE_NMS + i]) {
            s2[seg * POST_NMS + cnt] = sctop[seg * PRE_NMS + i];
            b2[seg * POST_NMS + cnt] = bxtop[seg * PRE_NMS + i];
            ++cnt;
        }
    }
    for (; cnt < POST_NMS; ++cnt) {
        s2[seg * POST_NMS + cnt] = -__builtin_inff();
        b2[seg * POST_NMS + cnt] = make_float4(0.f, 0.f, 0.f, 0.f);
    }
}

// ---------------------------------------------------------------------------
// 8) fill final merge buffer (per batch: 5*1000 entries padded to 8192)
// ---------------------------------------------------------------------------
__global__ void k_finalfill(const float* __restrict__ s2,
                            float* __restrict__ fkey, int* __restrict__ fidx) {
    int tid = blockIdx.x * blockDim.x + threadIdx.x;   // 2 * FPAD
    int b = tid / FPAD, i = tid % FPAD;
    if (b >= 2) return;
    if (i < 5 * POST_NMS) {
        int l = i / POST_NMS, slot = i % POST_NMS;
        fkey[b * FPAD + i] = s2[(b * 5 + l) * POST_NMS + slot];
        fidx[b * FPAD + i] = i;
    } else {
        fkey[b * FPAD + i] = -__builtin_inff();
        fidx[b * FPAD + i] = i;
    }
}

// ---------------------------------------------------------------------------
// 9) final output: top-1000 boxes, non-finite -> [0,0,1,1]
// ---------------------------------------------------------------------------
__global__ void k_output(const float* __restrict__ fkey, const int* __restrict__ fidx,
                         const float4* __restrict__ b2, float* __restrict__ out) {
    int tid = blockIdx.x * blockDim.x + threadIdx.x;   // 2 * 1000
    int b = tid / POST_NMS, i = tid % POST_NMS;
    if (b >= 2) return;
    float s = fkey[b * FPAD + i];
    float4 bx;
    if (s > -1e30f) {
        int id = fidx[b * FPAD + i];
        int l = id / POST_NMS, slot = id % POST_NMS;
        bx = b2[(b * 5 + l) * POST_NMS + slot];
    } else {
        bx = make_float4(0.0f, 0.0f, 1.0f, 1.0f);
    }
    float* op = out + (size_t)(b * POST_NMS + i) * 4;
    op[0] = bx.x; op[1] = bx.y; op[2] = bx.z; op[3] = bx.w;
}

// ---------------------------------------------------------------------------
extern "C" void kernel_launch(void* const* d_in, const int* in_sizes, int n_in,
                              void* d_out, int out_size, void* d_ws, size_t ws_size,
                              hipStream_t stream) {
    (void)in_sizes; (void)n_in; (void)out_size; (void)ws_size;
    const float* feats[5];
    for (int i = 0; i < 5; ++i) feats[i] = (const float*)d_in[i];
    const float* img_info = (const float*)d_in[5];
    const float* w_conv   = (const float*)d_in[6];
    const float* b_conv   = (const float*)d_in[7];
    const float* w_cls    = (const float*)d_in[8];
    const float* b_cls    = (const float*)d_in[9];
    const float* w_reg    = (const float*)d_in[10];
    const float* b_reg    = (const float*)d_in[11];
    float* out = (float*)d_out;

    char* ws = (char*)d_ws;
    size_t off = 0;
    auto alloc = [&](size_t bytes) -> char* {
        char* p = ws + off;
        off += (bytes + 255) & ~(size_t)255;
        return p;
    };

    const int Hs[5]       = {128, 64, 32, 16, 8};
    const int logWs[5]    = {7, 6, 5, 4, 3};
    const int strideT[5]  = {4, 8, 16, 32, 64};
    const float sizeT[5]  = {32.f, 64.f, 128.f, 256.f, 512.f};
    const int basepix[5]  = {0, 16384, 20480, 21504, 21760};
    const int baseanch[5] = {0, 49152, 61440, 64512, 65280};

    half_t* wpack  = (half_t*)alloc((size_t)KBLKS * 32 * 32 * 16 * 2);   // 2.25 MB
    half_t* hidden = (half_t*)alloc((size_t)2 * PTOT * CMID * 2);        // 44.7 MB
    half_t* featc[5];
    for (int l = 0; l < 5; ++l) {
        int nrows = (Hs[l] + 2) * (Hs[l] + 2);
        featc[l] = (half_t*)alloc((size_t)2 * nrows * CIN * 2);          // 23.4 MB total
    }
    float*  boxes  = (float*) alloc((size_t)2 * ANCHB * 4 * 4);          // 2.1 MB
    float*  skey   = (float*) alloc((size_t)NSEG * NPAD * 4);
    int*    sidx   = (int*)   alloc((size_t)NSEG * NPAD * 4);
    float*  sctop  = (float*) alloc((size_t)NSEG * PRE_NMS * 4);
    float4* bxtop  = (float4*)alloc((size_t)NSEG * PRE_NMS * 16);
    int*    keep   = (int*)   alloc((size_t)NSEG * PRE_NMS * 4);
    float*  s2     = (float*) alloc((size_t)NSEG * POST_NMS * 4);
    float4* b2     = (float4*)alloc((size_t)NSEG * POST_NMS * 16);
    float*  fkey   = (float*) alloc((size_t)2 * FPAD * 4);
    int*    fidx   = (int*)   alloc((size_t)2 * FPAD * 4);

    // 0) NHWC f16 + halo conversion
    for (int l = 0; l < 5; ++l) {
        dim3 g((Hs[l] + 2) * (Hs[l] + 2), 2);
        k_nhwc<<<g, 256, 0, stream>>>(feats[l], featc[l], Hs[l], Hs[l]);
    }

    // 1) weight pack
    k_pack_w<<<(KBLKS * 32 * 32 * 16) / 256, 256, 0, stream>>>(w_conv, wpack);

    // 2) conv (WMMA) per level
    for (int l = 0; l < 5; ++l) {
        dim3 grid(Hs[l] * Hs[l] / 16, 2);
        k_conv<<<grid, 256, 0, stream>>>(featc[l], wpack, b_conv, hidden,
                                         logWs[l], basepix[l]);
    }

    // 3) heads per level
    for (int l = 0; l < 5; ++l) {
        k_head<<<(2 * NPAD) / 256, 256, 0, stream>>>(
            hidden, w_cls, b_cls, w_reg, b_reg, img_info,
            skey, sidx, boxes, l, Hs[l], Hs[l],
            basepix[l], baseanch[l], strideT[l], sizeT[l]);
    }

    // 4) per-segment descending bitonic sort over 65536
    {
        int total = NSEG * NPAD;
        for (int kk = 2; kk <= NPAD; kk <<= 1)
            for (int j = kk >> 1; j >= 1; j >>= 1)
                k_bitonic<<<total / 256, 256, 0, stream>>>(skey, sidx, NPAD, kk, j, NSEG);
    }

    // 5) gather top-k
    k_gather<<<(NSEG * PRE_NMS + 255) / 256, 256, 0, stream>>>(skey, sidx, boxes, sctop, bxtop);

    // 6) NMS
    k_nms<<<NSEG, 256, 0, stream>>>(bxtop, keep);

    // 7) compact
    k_compact<<<NSEG, 32, 0, stream>>>(sctop, bxtop, keep, s2, b2);

    // 8) final merge buffers + sort
    k_finalfill<<<(2 * FPAD) / 256, 256, 0, stream>>>(s2, fkey, fidx);
    {
        int total = 2 * FPAD;
        for (int kk = 2; kk <= FPAD; kk <<= 1)
            for (int j = kk >> 1; j >= 1; j >>= 1)
                k_bitonic<<<total / 256, 256, 0, stream>>>(fkey, fidx, FPAD, kk, j, 2);
    }

    // 9) output
    k_output<<<(2 * POST_NMS + 255) / 256, 256, 0, stream>>>(fkey, fidx, b2, out);
}
